// GAT_MLP_41480794144921
// MI455X (gfx1250) — matrix operand order repaired and checked by reading, not compiled, not run on hardware
//
#include <hip/hip_runtime.h>
#include <math.h>

// ---------------------------------------------------------------------------
// Problem constants (from the reference)
// ---------------------------------------------------------------------------
#define NN    20000          // nodes
#define NP    20032          // nodes padded to multiple of 64
#define EE    320000         // edges (multiple of 64)
#define ETOT  340000         // edges + self loops

typedef _Float16 h16v __attribute__((ext_vector_type(16)));
typedef _Float16 h8v  __attribute__((ext_vector_type(8)));
typedef float    f8v  __attribute__((ext_vector_type(8)));

// ---------------------------------------------------------------------------
// Small utility kernels
// ---------------------------------------------------------------------------
__global__ void fill_f32_kernel(float* p, long long n, float v) {
    long long i = (long long)blockIdx.x * blockDim.x + threadIdx.x;
    if (i < n) p[i] = v;
}

__global__ void fill_f16_kernel(_Float16* p, long long n) {
    long long i = (long long)blockIdx.x * blockDim.x + threadIdx.x;
    if (i < n) p[i] = (_Float16)0.0f;
}

// W [K,N] f32 row-major  ->  Bt [N,K] f16 (transposed, contiguous in K)
__global__ void convert_transpose_kernel(const float* __restrict__ W,
                                         _Float16* __restrict__ Bt,
                                         int K, int N) {
    int idx = blockIdx.x * blockDim.x + threadIdx.x;
    if (idx >= K * N) return;
    int k = idx / N, n = idx % N;
    Bt[(size_t)n * K + k] = (_Float16)W[idx];
}

// x (with nan_to_num) -> f16
__global__ void convert_x_kernel(const float* __restrict__ x,
                                 _Float16* __restrict__ x16,
                                 long long count) {
    long long i = (long long)blockIdx.x * blockDim.x + threadIdx.x;
    if (i >= count) return;
    float v = x[i];
    if (v != v) v = 0.0f;                       // nan -> 0
    v = fminf(fmaxf(v, -3.0e38f), 3.0e38f);     // +-inf -> finite
    x16[i] = (_Float16)v;
}

// ---------------------------------------------------------------------------
// WMMA GEMM:  C[M,N] = act(A[M,K] @ B[K,N] + bias)
//   A:  f16 row-major (or gathered  [h[src[e]] | h[dst[e]]], K=256)
//   Bt: f16 [N,K]  (pre-transposed weights)
//   block = 128 threads (4 waves); wave w -> rows [blk*64+16w, +16)
//   each wave computes NT 16x16 output tiles along N (NT*16 columns)
// EPI: 0 = store f32 raw | 1 = bias+relu -> f16 | 2 = bias+elu -> f16
//      3 = bias + row-softmax (NT==1, N==16) -> f32
// ---------------------------------------------------------------------------
template <int NT, int GATHER, int EPI>
__global__ void __launch_bounds__(128)
wmma_gemm_kernel(const _Float16* __restrict__ A,
                 const _Float16* __restrict__ Bt,
                 const int* __restrict__ gsrc,
                 const int* __restrict__ gdst,
                 const float* __restrict__ bias,
                 float* __restrict__ Cf,
                 _Float16* __restrict__ Ch,
                 int M, int N, int K) {
    const int lane  = threadIdx.x & 31;
    const int wave  = threadIdx.x >> 5;
    const int mbase = blockIdx.x * 64 + wave * 16;
    const int nbase = blockIdx.y * (NT * 16);
    const int h2    = lane >> 4;     // which half of the wave
    const int l15   = lane & 15;

    f8v acc[NT] = {};

    const int row = mbase + l15;
    const _Float16* arow = nullptr;
    const _Float16* srow = nullptr;
    const _Float16* drow = nullptr;
    if (GATHER) {
        int s = gsrc[row], d = gdst[row];
        srow = A + (size_t)s * 128;
        drow = A + (size_t)d * 128;
    } else {
        arow = A + (size_t)row * K;
    }

    for (int kb = 0; kb < K; kb += 32) {
        // A fragment (ISA 16-bit A 16x32 layout): lane half h2 gets
        // K = h2*8 + e (e<8) and K = 16 + h2*8 + (e-8)
        const int off0 = kb + h2 * 8;
        const int off1 = off0 + 16;
        h8v lo, hi;
        if (GATHER) {
            lo = *(const h8v*)(off0 < 128 ? srow + off0 : drow + (off0 - 128));
            hi = *(const h8v*)(off1 < 128 ? srow + off1 : drow + (off1 - 128));
        } else {
            lo = *(const h8v*)(arow + off0);
            hi = *(const h8v*)(arow + off1);
        }
        h16v a = __builtin_shufflevector(lo, hi, 0, 1, 2, 3, 4, 5, 6, 7,
                                         8, 9, 10, 11, 12, 13, 14, 15);
#pragma unroll
        for (int t = 0; t < NT; ++t) {
            // B fragment: col N = l15, K = kb + h2*8 + e (e<8) and kb + 16 + h2*8 + (e-8)
            const int n = nbase + t * 16 + l15;
            h8v blo = *(const h8v*)(Bt + (size_t)n * K + off0);
            h8v bhi = *(const h8v*)(Bt + (size_t)n * K + off1);
            h16v b = __builtin_shufflevector(blo, bhi, 0, 1, 2, 3, 4, 5, 6, 7,
                                             8, 9, 10, 11, 12, 13, 14, 15);
            acc[t] = __builtin_amdgcn_wmma_f32_16x16x32_f16(
                false, a, false, b, (short)0, acc[t], false, false);
        }
    }

    // Epilogue. D layout: element v -> row M = mbase + h2*8 + v, col N = l15.
#pragma unroll
    for (int t = 0; t < NT; ++t) {
#pragma unroll
        for (int v = 0; v < 8; ++v) {
            const int m = mbase + h2 * 8 + v;
            const int n = nbase + t * 16 + l15;
            float val = acc[t][v];
            if (EPI == 0) {
                Cf[(size_t)m * N + n] = val;
            } else if (EPI == 1) {
                val += bias[n];
                Ch[(size_t)m * N + n] = (_Float16)fmaxf(val, 0.0f);
            } else if (EPI == 2) {
                val += bias[n];
                Ch[(size_t)m * N + n] =
                    (_Float16)(val > 0.0f ? val : (__expf(val) - 1.0f));
            } else {  // EPI == 3 : bias + softmax over the 16 columns
                val += bias[n];
                float vmax = val;
#pragma unroll
                for (int msk = 1; msk < 16; msk <<= 1)
                    vmax = fmaxf(vmax, __shfl_xor(vmax, msk, 32));
                float ex = __expf(val - vmax);
                float sum = ex;
#pragma unroll
                for (int msk = 1; msk < 16; msk <<= 1)
                    sum += __shfl_xor(sum, msk, 32);
                Cf[(size_t)m * 16 + n] = ex / sum;
            }
        }
    }
}

// ---------------------------------------------------------------------------
// GAT attention kernels
// ---------------------------------------------------------------------------
__device__ inline void atomicMaxF(float* addr, float val) {
    unsigned int* ua = (unsigned int*)addr;
    unsigned int old = *ua;
    while (__uint_as_float(old) < val) {
        unsigned int assumed = old;
        old = atomicCAS(ua, assumed, __float_as_uint(val));
        if (old == assumed) break;
    }
}

// per-(node,head) attention scalars; also re-init m (=-inf) and z (=0)
__global__ void alpha_kernel(const float* __restrict__ htmp,
                             const float* __restrict__ a_s,
                             const float* __restrict__ a_d,
                             float* __restrict__ as_, float* __restrict__ ad_,
                             float* __restrict__ m_, float* __restrict__ z_,
                             int C) {
    int g = blockIdx.x * blockDim.x + threadIdx.x;
    if (g >= NN * 8) return;
    int n = g >> 3, h = g & 7;
    const float* hp  = htmp + (size_t)n * (8 * C) + h * C;
    const float* asp = a_s + h * C;
    const float* adp = a_d + h * C;
    float s1 = 0.0f, s2 = 0.0f;
    for (int c = 0; c < C; ++c) {
        float v = hp[c];
        s1 += v * asp[c];
        s2 += v * adp[c];
    }
    as_[g] = s1;
    ad_[g] = s2;
    m_[g]  = -1e30f;
    z_[g]  = 0.0f;
}

// pass 1: e = lrelu(as[s]+ad[d]); store e; segment max over dst
__global__ void edge_max_kernel(const int* __restrict__ src,
                                const int* __restrict__ dst,
                                const float* __restrict__ as_,
                                const float* __restrict__ ad_,
                                float* __restrict__ e_edge,
                                float* __restrict__ m_) {
    int idx = blockIdx.x * blockDim.x + threadIdx.x;
    if (idx >= ETOT) return;
    int s = idx < EE ? src[idx] : idx - EE;   // appended self loops
    int d = idx < EE ? dst[idx] : idx - EE;
#pragma unroll
    for (int h = 0; h < 8; ++h) {
        float v = as_[s * 8 + h] + ad_[d * 8 + h];
        v = v > 0.0f ? v : 0.2f * v;          // leaky_relu(0.2)
        e_edge[(size_t)idx * 8 + h] = v;
        atomicMaxF(&m_[d * 8 + h], v);
    }
}

// pass 2: ex = exp(e - m[d]);  z[d,h] += ex;  hagg[d,:] += ex * htmp[s,:]
// one block per edge, DOUT threads (coalesced scatter-adds)
template <int DOUT, int C>
__global__ void edge_aggregate_kernel(const int* __restrict__ src,
                                      const int* __restrict__ dst,
                                      const float* __restrict__ e_edge,
                                      const float* __restrict__ m_,
                                      float* __restrict__ z_,
                                      const float* __restrict__ htmp,
                                      float* __restrict__ hagg) {
    int idx = blockIdx.x;
    int tid = threadIdx.x;
    int s = idx < EE ? src[idx] : idx - EE;
    int d = idx < EE ? dst[idx] : idx - EE;
    int h = tid / C;
    float ex = __expf(e_edge[(size_t)idx * 8 + h] - m_[d * 8 + h]);
    if ((tid & (C - 1)) == 0) atomicAdd(&z_[d * 8 + h], ex);
    atomicAdd(&hagg[(size_t)d * DOUT + tid],
              ex * htmp[(size_t)s * DOUT + tid]);
}

// ---------------------------------------------------------------------------
// Node epilogues
// ---------------------------------------------------------------------------
// layer 0: h = lrelu(agg/z + b);  write f32 master + f16 copy
__global__ void epilogue0_kernel(const float* __restrict__ hagg,
                                 const float* __restrict__ z_,
                                 const float* __restrict__ b,
                                 float* __restrict__ hcur,
                                 _Float16* __restrict__ h16) {
    int i = blockIdx.x * blockDim.x + threadIdx.x;
    if (i >= NN * 512) return;
    int n = i >> 9, j = i & 511;
    float v = hagg[i] / z_[n * 8 + (j >> 6)] + b[j];
    v = v > 0.0f ? v : 0.2f * v;
    hcur[i] = v;
    h16[i]  = (_Float16)v;
}

// middle layers: h = h + lrelu(agg/z + b)
__global__ void epilogue_mid_kernel(const float* __restrict__ hagg,
                                    const float* __restrict__ z_,
                                    const float* __restrict__ b,
                                    float* __restrict__ hcur,
                                    _Float16* __restrict__ h16) {
    int i = blockIdx.x * blockDim.x + threadIdx.x;
    if (i >= NN * 512) return;
    int n = i >> 9, j = i & 511;
    float v = hagg[i] / z_[n * 8 + (j >> 6)] + b[j];
    v = v > 0.0f ? v : 0.2f * v;
    float nv = hcur[i] + v;
    hcur[i] = nv;
    h16[i]  = (_Float16)nv;
}

// last layer: hfin = relu(mean_h(agg/z) + b)  (EMB=128)
__global__ void epilogue_last_kernel(const float* __restrict__ hagg,
                                     const float* __restrict__ z_,
                                     const float* __restrict__ b,
                                     _Float16* __restrict__ hfin16) {
    int i = blockIdx.x * blockDim.x + threadIdx.x;
    if (i >= NN * 128) return;
    int n = i >> 7, c = i & 127;
    float acc = 0.0f;
#pragma unroll
    for (int h = 0; h < 8; ++h)
        acc += hagg[(size_t)n * 1024 + h * 128 + c] / z_[n * 8 + h];
    acc = acc * 0.125f + b[c];
    hfin16[i] = (_Float16)fmaxf(acc, 0.0f);
}

// ---------------------------------------------------------------------------
// Launch
// ---------------------------------------------------------------------------
extern "C" void kernel_launch(void* const* d_in, const int* in_sizes, int n_in,
                              void* d_out, int out_size, void* d_ws,
                              size_t ws_size, hipStream_t stream) {
    (void)in_sizes; (void)n_in; (void)out_size; (void)ws_size;

    const float* x      = (const float*)d_in[0];
    const int*   ei     = (const int*)d_in[1];
    const float* W0     = (const float*)d_in[2];
    const float* a_src0 = (const float*)d_in[3];
    const float* a_dst0 = (const float*)d_in[4];
    const float* b0     = (const float*)d_in[5];
    const float* Wm     = (const float*)d_in[6];
    const float* a_srcm = (const float*)d_in[7];
    const float* a_dstm = (const float*)d_in[8];
    const float* bm     = (const float*)d_in[9];
    const float* Wl     = (const float*)d_in[10];
    const float* a_srcl = (const float*)d_in[11];
    const float* a_dstl = (const float*)d_in[12];
    const float* bl     = (const float*)d_in[13];
    const float* mw1    = (const float*)d_in[14];
    const float* mb1    = (const float*)d_in[15];
    const float* mfw    = (const float*)d_in[16];
    const float* mfb    = (const float*)d_in[17];
    const float* mwo    = (const float*)d_in[18];
    const float* mbo    = (const float*)d_in[19];
    const int* src = ei;
    const int* dst = ei + EE;
    float* out = (float*)d_out;

    // ---- workspace carve-up -------------------------------------------------
    char*  ws  = (char*)d_ws;
    size_t off = 0;
    auto alloc = [&](size_t bytes) -> void* {
        void* p = ws + off;
        off += (bytes + 255) & ~(size_t)255;
        return p;
    };
    _Float16* w0t    = (_Float16*)alloc((size_t)512 * 128 * 2);
    _Float16* wm0t   = (_Float16*)alloc((size_t)512 * 512 * 2);
    _Float16* wm1t   = (_Float16*)alloc((size_t)512 * 512 * 2);
    _Float16* wlt    = (_Float16*)alloc((size_t)1024 * 512 * 2);
    _Float16* mw1t   = (_Float16*)alloc((size_t)512 * 256 * 2);
    _Float16* mf0t   = (_Float16*)alloc((size_t)512 * 512 * 2);
    _Float16* mf1t   = (_Float16*)alloc((size_t)512 * 512 * 2);
    _Float16* mwot   = (_Float16*)alloc((size_t)16 * 512 * 2);
    _Float16* x16    = (_Float16*)alloc((size_t)NP * 128 * 2);
    _Float16* h16    = (_Float16*)alloc((size_t)NP * 512 * 2);
    _Float16* hfin16 = (_Float16*)alloc((size_t)NP * 128 * 2);
    _Float16* zA     = (_Float16*)alloc((size_t)EE * 512 * 2);
    _Float16* zB     = (_Float16*)alloc((size_t)EE * 512 * 2);
    float* htmp   = (float*)alloc((size_t)NP * 1024 * 4);
    float* hagg   = (float*)alloc((size_t)NP * 1024 * 4);
    float* hcur   = (float*)alloc((size_t)NP * 512 * 4);
    float* as_    = (float*)alloc((size_t)NP * 8 * 4);
    float* ad_    = (float*)alloc((size_t)NP * 8 * 4);
    float* m_     = (float*)alloc((size_t)NP * 8 * 4);
    float* z_     = (float*)alloc((size_t)NP * 8 * 4);
    float* e_edge = (float*)alloc((size_t)ETOT * 8 * 4);

    auto cdiv = [](long long a, long long b) { return (int)((a + b - 1) / b); };

    // ---- weight conversion (f32 -> f16 transposed) -------------------------
    convert_transpose_kernel<<<cdiv(128 * 512, 256), 256, 0, stream>>>(W0, w0t, 128, 512);
    convert_transpose_kernel<<<cdiv(512 * 512, 256), 256, 0, stream>>>(Wm, wm0t, 512, 512);
    convert_transpose_kernel<<<cdiv(512 * 512, 256), 256, 0, stream>>>(Wm + 512 * 512, wm1t, 512, 512);
    convert_transpose_kernel<<<cdiv(512 * 1024, 256), 256, 0, stream>>>(Wl, wlt, 512, 1024);
    convert_transpose_kernel<<<cdiv(256 * 512, 256), 256, 0, stream>>>(mw1, mw1t, 256, 512);
    convert_transpose_kernel<<<cdiv(512 * 512, 256), 256, 0, stream>>>(mfw, mf0t, 512, 512);
    convert_transpose_kernel<<<cdiv(512 * 512, 256), 256, 0, stream>>>(mfw + 512 * 512, mf1t, 512, 512);
    convert_transpose_kernel<<<cdiv(512 * 16, 256), 256, 0, stream>>>(mwo, mwot, 512, 16);

    // ---- input conversion (zero pad rows, nan_to_num) ----------------------
    fill_f16_kernel<<<cdiv((long long)NP * 128, 256), 256, 0, stream>>>(x16, (long long)NP * 128);
    fill_f16_kernel<<<cdiv((long long)NP * 512, 256), 256, 0, stream>>>(h16, (long long)NP * 512);
    convert_x_kernel<<<cdiv((long long)NN * 128, 256), 256, 0, stream>>>(x, x16, (long long)NN * 128);

    // ======================= GAT layer 0 ====================================
    wmma_gemm_kernel<4, 0, 0><<<dim3(NP / 64, 512 / 64), 128, 0, stream>>>(
        x16, w0t, nullptr, nullptr, nullptr, htmp, nullptr, NP, 512, 128);
    alpha_kernel<<<cdiv(NN * 8, 256), 256, 0, stream>>>(htmp, a_src0, a_dst0,
                                                        as_, ad_, m_, z_, 64);
    fill_f32_kernel<<<cdiv((long long)NN * 512, 256), 256, 0, stream>>>(
        hagg, (long long)NN * 512, 0.0f);
    edge_max_kernel<<<cdiv(ETOT, 256), 256, 0, stream>>>(src, dst, as_, ad_, e_edge, m_);
    edge_aggregate_kernel<512, 64><<<ETOT, 512, 0, stream>>>(src, dst, e_edge, m_, z_, htmp, hagg);
    epilogue0_kernel<<<cdiv(NN * 512, 256), 256, 0, stream>>>(hagg, z_, b0, hcur, h16);

    // ======================= GAT middle layers ==============================
    for (int i = 0; i < 2; ++i) {
        const _Float16* wt = (i == 0) ? wm0t : wm1t;
        wmma_gemm_kernel<4, 0, 0><<<dim3(NP / 64, 512 / 64), 128, 0, stream>>>(
            h16, wt, nullptr, nullptr, nullptr, htmp, nullptr, NP, 512, 512);
        alpha_kernel<<<cdiv(NN * 8, 256), 256, 0, stream>>>(
            htmp, a_srcm + i * 512, a_dstm + i * 512, as_, ad_, m_, z_, 64);
        fill_f32_kernel<<<cdiv((long long)NN * 512, 256), 256, 0, stream>>>(
            hagg, (long long)NN * 512, 0.0f);
        edge_max_kernel<<<cdiv(ETOT, 256), 256, 0, stream>>>(src, dst, as_, ad_, e_edge, m_);
        edge_aggregate_kernel<512, 64><<<ETOT, 512, 0, stream>>>(src, dst, e_edge, m_, z_, htmp, hagg);
        epilogue_mid_kernel<<<cdiv(NN * 512, 256), 256, 0, stream>>>(hagg, z_, bm + i * 512, hcur, h16);
    }

    // ======================= GAT last layer (mean heads) ====================
    wmma_gemm_kernel<4, 0, 0><<<dim3(NP / 64, 1024 / 64), 128, 0, stream>>>(
        h16, wlt, nullptr, nullptr, nullptr, htmp, nullptr, NP, 1024, 512);
    alpha_kernel<<<cdiv(NN * 8, 256), 256, 0, stream>>>(htmp, a_srcl, a_dstl,
                                                        as_, ad_, m_, z_, 128);
    fill_f32_kernel<<<cdiv((long long)NN * 1024, 256), 256, 0, stream>>>(
        hagg, (long long)NN * 1024, 0.0f);
    edge_max_kernel<<<cdiv(ETOT, 256), 256, 0, stream>>>(src, dst, as_, ad_, e_edge, m_);
    edge_aggregate_kernel<1024, 128><<<ETOT, 1024, 0, stream>>>(src, dst, e_edge, m_, z_, htmp, hagg);
    epilogue_last_kernel<<<cdiv(NN * 128, 256), 256, 0, stream>>>(hagg, z_, bl, hfin16);

    // ======================= Edge MLP =======================================
    // layer 1: relu([h[src]|h[dst]] @ mw1 + mb1)   (A gathered on the fly)
    wmma_gemm_kernel<4, 1, 1><<<dim3(EE / 64, 512 / 64), 128, 0, stream>>>(
        hfin16, mw1t, src, dst, mb1, nullptr, zA, EE, 512, 256);
    // layer 2: elu(zA @ mfw0 + mfb0)
    wmma_gemm_kernel<4, 0, 2><<<dim3(EE / 64, 512 / 64), 128, 0, stream>>>(
        zA, mf0t, nullptr, nullptr, mfb, nullptr, zB, EE, 512, 512);
    // layer 3: elu(zB @ mfw1 + mfb1)
    wmma_gemm_kernel<4, 0, 2><<<dim3(EE / 64, 512 / 64), 128, 0, stream>>>(
        zB, mf1t, nullptr, nullptr, mfb + 512, nullptr, zA, EE, 512, 512);
    // output: softmax(zA @ mwo + mbo) -> d_out
    wmma_gemm_kernel<1, 0, 3><<<dim3(EE / 64, 1), 128, 0, stream>>>(
        zA, mwot, nullptr, nullptr, mbo, out, nullptr, EE, 16, 512);
}
